// SelfAttention_41188736368831
// MI455X (gfx1250) — matrix operand order repaired and checked by reading, not compiled, not run on hardware
//
#include <hip/hip_runtime.h>

// ---------------------------------------------------------------------------
// Self-attention for MI455X (gfx1250): bf16 WMMA everywhere, flash-attention
// style online softmax, TDM (tensor_load_to_lds) double-buffered GEMM tiles.
// ---------------------------------------------------------------------------

#define TSEQ  2048
#define CDIM  1024
#define BATCH 4
#define HEADS 16
#define DH    64

typedef __bf16 bf16_t;
typedef bf16_t v16bf __attribute__((ext_vector_type(16)));
typedef float  v8f   __attribute__((ext_vector_type(8)));
typedef unsigned int v4u __attribute__((ext_vector_type(4)));
typedef int          v8i __attribute__((ext_vector_type(8)));
typedef int          v4i __attribute__((ext_vector_type(4)));

#define HAVE_TDM __has_builtin(__builtin_amdgcn_tensor_load_to_lds)

struct BFrag { uint4 lo, hi; };   // 32 bytes == one 16-element bf16 fragment

__device__ __forceinline__ v16bf make_frag(const unsigned short* lo,
                                           const unsigned short* hi) {
  BFrag f;
  f.lo = *reinterpret_cast<const uint4*>(lo);
  f.hi = *reinterpret_cast<const uint4*>(hi);
  return __builtin_bit_cast(v16bf, f);
}

__device__ __forceinline__ unsigned short f32_to_bf16u(float x) {
  unsigned int u = __builtin_bit_cast(unsigned int, x);
  u += 0x7FFFu + ((u >> 16) & 1u);           // round-to-nearest-even
  return (unsigned short)(u >> 16);
}

__device__ __forceinline__ v8f wmma_bf16(v16bf a, v16bf b, v8f c) {
  // D = A(16x32) * B(32x16) + C, fp32 accumulate
  return __builtin_amdgcn_wmma_f32_16x16x32_bf16(false, a, false, b,
                                                 (short)0, c, false, false);
}

#if HAVE_TDM
// ---------------------------------------------------------------------------
// TDM 2D tile load: global [tile_d1 rows x tile_d0 elems] (2-byte elements,
// row stride = stride_elems) -> contiguous LDS at lds_addr. Descriptor per
// CDNA5 ISA ch.8 (D# group0/group1; groups 2/3 zero for 2D).
// 6-arg builtin form (clang-23 / therock-10.0 headers).
// ---------------------------------------------------------------------------
__device__ __forceinline__ void tdm_load_2d(unsigned lds_addr,
                                            const unsigned short* gptr,
                                            unsigned tile_d0, unsigned tile_d1,
                                            unsigned stride_elems) {
  unsigned long long ga = (unsigned long long)(size_t)gptr;
  const unsigned td0 = 1u << 20, td1 = 1u << 20;   // generous in-bounds dims
  v4u g0;
  g0.x = 1u;                                        // count=1 (valid user D#)
  g0.y = lds_addr;                                  // LDS byte address
  g0.z = (unsigned)ga;                              // global_addr[31:0]
  g0.w = (unsigned)((ga >> 32) & 0x01FFFFFFu) | (2u << 30);  // addr[56:32]|type=2
  v8i g1;
  g1[0] = (int)(1u << 16);                          // wg_mask=0, data_size=1 (2B)
  g1[1] = (int)((td0 & 0xFFFFu) << 16);             // tensor_dim0 lo
  g1[2] = (int)((td0 >> 16) | ((td1 & 0xFFFFu) << 16));
  g1[3] = (int)((td1 >> 16) | (tile_d0 << 16));     // tile_dim0
  g1[4] = (int)tile_d1;                             // tile_dim1 (tile_dim2=0)
  g1[5] = (int)stride_elems;                        // tensor_dim0_stride lo
  g1[6] = 0;                                        // stride hi / dim1 stride
  g1[7] = 0;
  v4i z4 = {0, 0, 0, 0};
  v8i z8 = {0, 0, 0, 0, 0, 0, 0, 0};
  __builtin_amdgcn_tensor_load_to_lds(g0, g1, z4, z4, z8, 0);
}
#endif

// ---------------------------------------------------------------------------
// fp32 -> bf16 bulk convert (8 elements / thread, 16B stores)
// ---------------------------------------------------------------------------
__global__ __launch_bounds__(256) void cvt_f32_bf16_kernel(
    const float* __restrict__ in, unsigned short* __restrict__ out, int n8) {
  int i = blockIdx.x * 256 + threadIdx.x;
  if (i >= n8) return;
  const float4* p = reinterpret_cast<const float4*>(in) + (size_t)i * 2;
  float4 a = p[0];
  float4 b = p[1];
  union { uint4 u; unsigned short s[8]; } pk;
  pk.s[0] = f32_to_bf16u(a.x); pk.s[1] = f32_to_bf16u(a.y);
  pk.s[2] = f32_to_bf16u(a.z); pk.s[3] = f32_to_bf16u(a.w);
  pk.s[4] = f32_to_bf16u(b.x); pk.s[5] = f32_to_bf16u(b.y);
  pk.s[6] = f32_to_bf16u(b.z); pk.s[7] = f32_to_bf16u(b.w);
  reinterpret_cast<uint4*>(out)[i] = pk.u;
}

// ---------------------------------------------------------------------------
// W[K,N] fp32  ->  Wt[N,K] bf16 (tiled transpose through LDS)
// ---------------------------------------------------------------------------
__global__ __launch_bounds__(256) void cvt_transpose_kernel(
    const float* __restrict__ W, unsigned short* __restrict__ Wt,
    int Kd, int N) {
  __shared__ float tile[32][33];
  const int n0 = blockIdx.x * 32;
  const int k0 = blockIdx.y * 32;
  const int tx = threadIdx.x & 31;
  const int ty = threadIdx.x >> 5;
#pragma unroll
  for (int i = ty; i < 32; i += 8)
    tile[i][tx] = W[(size_t)(k0 + i) * N + (n0 + tx)];
  __syncthreads();
#pragma unroll
  for (int i = ty; i < 32; i += 8)
    Wt[(size_t)(n0 + i) * Kd + (k0 + tx)] = f32_to_bf16u(tile[tx][i]);
}

// ---------------------------------------------------------------------------
// Shared GEMM main loop: C[128x128] += A[128xK] * Bt[128xK]^T
// 8 waves, each wave computes a 32x64 sub-tile = 2x4 WMMA accumulators.
// Tiles are 128x64 bf16, double-buffered, moved by the Tensor Data Mover
// (each wave DMAs its own 16-row slab of A and Bt; TENSORcnt + barrier sync).
// ---------------------------------------------------------------------------
__device__ __forceinline__ void gemm_tile_mainloop(
    const unsigned short* __restrict__ A,
    const unsigned short* __restrict__ Bt,
    int K, int mBase, int nBase,
    unsigned short* lds_a, unsigned short* lds_b, v8f (&acc)[2][4]) {
  const int tid   = threadIdx.x;
  const int lane  = tid & 31;
  const int l16   = lane & 15;
  const int khalf = lane >> 4;
  const int wid   = tid >> 5;
  const int wm = (wid & 3) * 32;   // wave row offset
  const int wn = (wid >> 2) * 64;  // wave col offset

#if HAVE_TDM
  const int uwid = __builtin_amdgcn_readfirstlane(wid);
  const unsigned ldsA0 = (unsigned)(size_t)lds_a;   // LDS aperture low bits
  const unsigned ldsB0 = (unsigned)(size_t)lds_b;
  auto issue = [&](int k0, int buf) {
    const unsigned off = (unsigned)(buf * 16384 + uwid * 2048);  // bytes
    tdm_load_2d(ldsA0 + off, A + (size_t)(mBase + uwid * 16) * K + k0,
                64u, 16u, (unsigned)K);
    tdm_load_2d(ldsB0 + off, Bt + (size_t)(nBase + uwid * 16) * K + k0,
                64u, 16u, (unsigned)K);
  };
  const int iters = K / 64;
  issue(0, 0);
  for (int it = 0; it < iters; ++it) {
    const int cur = it & 1;
    __syncthreads();                 // everyone done computing on buf cur^1
    if (it + 1 < iters) {
      issue((it + 1) * 64, cur ^ 1); // prefetch next tile via TDM
      __builtin_amdgcn_s_wait_tensorcnt(2);   // my current slabs arrived
    } else {
      __builtin_amdgcn_s_wait_tensorcnt(0);
    }
    __syncthreads();                 // all waves' slabs of buf cur arrived
    const unsigned short* la = lds_a + cur * 8192;
    const unsigned short* lb = lds_b + cur * 8192;
#else
  for (int k0 = 0; k0 < K; k0 += 64) {
    // Fallback staging: 4 x 16B per tile per thread
#pragma unroll
    for (int u = tid; u < 1024; u += 256) {
      const int row = u >> 3, seg = u & 7;
      reinterpret_cast<uint4*>(lds_a)[u] =
          *reinterpret_cast<const uint4*>(A + (size_t)(mBase + row) * K + k0 + seg * 8);
      reinterpret_cast<uint4*>(lds_b)[u] =
          *reinterpret_cast<const uint4*>(Bt + (size_t)(nBase + row) * K + k0 + seg * 8);
    }
    __syncthreads();
    const unsigned short* la = lds_a;
    const unsigned short* lb = lds_b;
#endif
#pragma unroll
    for (int c = 0; c < 2; ++c) {   // two K=32 chunks
      v16bf af[2], bfr[4];
#pragma unroll
      for (int r = 0; r < 2; ++r) {
        // A frag: row = l16, K run [khalf*16,+16)
        const unsigned short* p = la + (wm + r * 16 + l16) * 64 + c * 32;
        af[r] = make_frag(p + khalf * 16, p + khalf * 16 + 8);
      }
#pragma unroll
      for (int j = 0; j < 4; ++j) {
        // B frag: col = l16, K run [khalf*16,+16) (contiguous, Bt is [N,K])
        const unsigned short* p = lb + (wn + j * 16 + l16) * 64 + c * 32 + khalf * 16;
        bfr[j] = make_frag(p, p + 8);
      }
#pragma unroll
      for (int r = 0; r < 2; ++r)
#pragma unroll
        for (int j = 0; j < 4; ++j)
          acc[r][j] = wmma_bf16(af[r], bfr[j], acc[r][j]);
    }
#if !HAVE_TDM
    __syncthreads();
#endif
  }
}

// ---------------------------------------------------------------------------
// QKV projection: xb[8192,1024] @ Wqkv -> scatter Q[BH,T,64] K[BH,T,64]
// Vt[BH,64,T] (V pre-transposed so the attention P@V B-operand is contiguous)
// ---------------------------------------------------------------------------
__global__ __launch_bounds__(256) void gemm_qkv_kernel(
    const unsigned short* __restrict__ Xb,
    const unsigned short* __restrict__ Wt,
    const float* __restrict__ bias,
    unsigned short* __restrict__ Qo,
    unsigned short* __restrict__ Ko,
    unsigned short* __restrict__ Vto) {
  __shared__ __align__(16) unsigned short lds_a[2 * 128 * 64];
  __shared__ __align__(16) unsigned short lds_b[2 * 128 * 64];
  const int mBase = blockIdx.y * 128;
  const int nBase = blockIdx.x * 128;
  v8f acc[2][4] = {};
  gemm_tile_mainloop(Xb, Wt, CDIM, mBase, nBase, lds_a, lds_b, acc);

  const int lane  = threadIdx.x & 31;
  const int wid   = threadIdx.x >> 5;
  const int l16   = lane & 15;
  const int khalf = lane >> 4;
  const int wm = (wid & 3) * 32;
  const int wn = (wid >> 2) * 64;
#pragma unroll
  for (int r = 0; r < 2; ++r)
#pragma unroll
    for (int j = 0; j < 4; ++j)
#pragma unroll
      for (int v = 0; v < 8; ++v) {
        const int m = mBase + wm + r * 16 + v + 8 * khalf;
        const int n = nBase + wn + j * 16 + l16;
        const unsigned short bv = f32_to_bf16u(acc[r][j][v] + bias[n]);
        const int b = m >> 11, t = m & 2047;            // m = b*T + t
        const int which = n >> 10, cc = n & 1023;       // q/k/v select
        const int h = cc >> 6, d = cc & 63;
        const size_t bh = (size_t)(b * HEADS + h);
        if (which == 0)      Qo[(bh * TSEQ + t) * DH + d] = bv;
        else if (which == 1) Ko[(bh * TSEQ + t) * DH + d] = bv;
        else                 Vto[(bh * DH + d) * TSEQ + t] = bv;
      }
}

// ---------------------------------------------------------------------------
// Flash attention: block = 128 queries of one (b,h); wave = 16 queries.
// Per 32-key block: 4 WMMA (S = Q K^T), online softmax, 4 WMMA (O += P V).
// ---------------------------------------------------------------------------
__global__ __launch_bounds__(256) void attn_kernel(
    const unsigned short* __restrict__ Q,
    const unsigned short* __restrict__ Kmat,
    const unsigned short* __restrict__ Vt,
    const unsigned char* __restrict__ mask,
    unsigned short* __restrict__ Yb) {
  __shared__ __align__(16) unsigned short lds_p[8][16 * 32];  // per-wave P tile

  const int tid   = threadIdx.x;
  const int wid   = tid >> 5;
  const int lane  = tid & 31;
  const int l16   = lane & 15;
  const int khalf = lane >> 4;

  const int bh = blockIdx.y;       // b*H + h
  const int b  = bh >> 4;
  const int h  = bh & 15;
  const int qbase = blockIdx.x * 128 + wid * 16;

  const unsigned short* Qb = Q    + (size_t)bh * TSEQ * DH;
  const unsigned short* Kb = Kmat + (size_t)bh * TSEQ * DH;
  const unsigned short* Vb = Vt   + (size_t)bh * DH * TSEQ;
  const unsigned char*  mb = mask + (size_t)b * TSEQ;

  // Q A-fragments over Dh=64 (two K=32 chunks); loaded once, reused 64x
  v16bf qf[2];
  {
    const unsigned short* qr = Qb + (size_t)(qbase + l16) * DH;
#pragma unroll
    for (int c = 0; c < 2; ++c)
      qf[c] = make_frag(qr + c * 32 + khalf * 16, qr + c * 32 + khalf * 16 + 8);
  }

  bool maskq[8];
#pragma unroll
  for (int v = 0; v < 8; ++v) maskq[v] = mb[qbase + v + 8 * khalf] != 0;

  float mrow[8], lrow[8];
#pragma unroll
  for (int v = 0; v < 8; ++v) { mrow[v] = -3.0e38f; lrow[v] = 0.0f; }
  v8f acc[4] = {};                 // O accumulator: 16 queries x 64 dims
  const float scale = 0.125f;      // 1/sqrt(64)

  for (int kb = 0; kb < TSEQ; kb += 32) {
    if (kb + 32 < TSEQ)            // global_prefetch next K rows
      __builtin_prefetch(Kb + (size_t)(kb + 32 + lane) * DH, 0, 1);
    // ---- S = Q K^T for two 16-key groups -------------------------------
    v8f s[2];
    bool mk[2];
#pragma unroll
    for (int g = 0; g < 2; ++g) {
      const int key = kb + g * 16 + l16;
      const unsigned short* kr = Kb + (size_t)key * DH + khalf * 16;
      v16bf kf0 = make_frag(kr, kr + 8);            // d in [0,32)
      v16bf kf1 = make_frag(kr + 32, kr + 40);      // d in [32,64)
      v8f z = {};
      v8f t0 = wmma_bf16(qf[0], kf0, z);
      s[g]   = wmma_bf16(qf[1], kf1, t0);
      mk[g]  = mb[key] != 0;
    }
    // ---- scale + padding mask (column 0 always allowed) ----------------
    float sv[2][8];
#pragma unroll
    for (int g = 0; g < 2; ++g) {
      const int key = kb + g * 16 + l16;
#pragma unroll
      for (int v = 0; v < 8; ++v) {
        const bool ok = (key == 0) || (maskq[v] && mk[g]);
        sv[g][v] = ok ? s[g][v] * scale : -1.0e30f;
      }
    }
    // ---- online softmax (row = 16-lane half, per D-layout) -------------
    float pv[2][8];
#pragma unroll
    for (int v = 0; v < 8; ++v) {
      float t = fmaxf(sv[0][v], sv[1][v]);
#pragma unroll
      for (int off = 1; off < 16; off <<= 1)
        t = fmaxf(t, __shfl_xor(t, off, 32));
      const float mnew  = fmaxf(mrow[v], t);
      const float alpha = __expf(mrow[v] - mnew);
      const float p0 = __expf(sv[0][v] - mnew);
      const float p1 = __expf(sv[1][v] - mnew);
      pv[0][v] = p0; pv[1][v] = p1;
      float rs = p0 + p1;
#pragma unroll
      for (int off = 1; off < 16; off <<= 1)
        rs += __shfl_xor(rs, off, 32);
      lrow[v] = lrow[v] * alpha + rs;
      mrow[v] = mnew;
#pragma unroll
      for (int j = 0; j < 4; ++j) acc[j][v] *= alpha;
    }
    // ---- D-layout -> A-layout via per-wave LDS (no barrier needed) -----
    unsigned short* pw = &lds_p[wid][0];
#pragma unroll
    for (int g = 0; g < 2; ++g)
#pragma unroll
      for (int v = 0; v < 8; ++v)
        pw[(v + 8 * khalf) * 32 + g * 16 + l16] = f32_to_bf16u(pv[g][v]);
    const unsigned short* pr = pw + l16 * 32;
    v16bf pf = make_frag(pr + khalf * 16, pr + khalf * 16 + 8);
    // ---- O += P @ V  (Vt is [Dh, T] so key-runs are contiguous) --------
#pragma unroll
    for (int j = 0; j < 4; ++j) {
      const unsigned short* vr =
          Vb + (size_t)(j * 16 + l16) * TSEQ + kb + khalf * 16;
      v16bf vf = make_frag(vr, vr + 8);
      acc[j] = wmma_bf16(pf, vf, acc[j]);
    }
  }

  // ---- normalize, write y as bf16 [B*T, C] -----------------------------
#pragma unroll
  for (int v = 0; v < 8; ++v) {
    const float inv = 1.0f / lrow[v];
    const int q = qbase + v + 8 * khalf;
#pragma unroll
    for (int j = 0; j < 4; ++j) {
      const int col = h * DH + j * 16 + l16;
      Yb[(size_t)(b * TSEQ + q) * CDIM + col] = f32_to_bf16u(acc[j][v] * inv);
    }
  }
}

// ---------------------------------------------------------------------------
// Output projection: yb[8192,1024] @ Wout -> out fp32 (+ bias)
// ---------------------------------------------------------------------------
__global__ __launch_bounds__(256) void gemm_out_kernel(
    const unsigned short* __restrict__ Yb,
    const unsigned short* __restrict__ Wot,
    const float* __restrict__ bias,
    float* __restrict__ out) {
  __shared__ __align__(16) unsigned short lds_a[2 * 128 * 64];
  __shared__ __align__(16) unsigned short lds_b[2 * 128 * 64];
  const int mBase = blockIdx.y * 128;
  const int nBase = blockIdx.x * 128;
  v8f acc[2][4] = {};
  gemm_tile_mainloop(Yb, Wot, CDIM, mBase, nBase, lds_a, lds_b, acc);

  const int lane  = threadIdx.x & 31;
  const int wid   = threadIdx.x >> 5;
  const int l16   = lane & 15;
  const int khalf = lane >> 4;
  const int wm = (wid & 3) * 32;
  const int wn = (wid >> 2) * 64;
#pragma unroll
  for (int r = 0; r < 2; ++r)
#pragma unroll
    for (int j = 0; j < 4; ++j)
#pragma unroll
      for (int v = 0; v < 8; ++v) {
        const int m = mBase + wm + r * 16 + v + 8 * khalf;
        const int n = nBase + wn + j * 16 + l16;
        out[(size_t)m * CDIM + n] = acc[r][j][v] + bias[n];
      }
}

// ---------------------------------------------------------------------------
extern "C" void kernel_launch(void* const* d_in, const int* in_sizes, int n_in,
                              void* d_out, int out_size, void* d_ws, size_t ws_size,
                              hipStream_t stream) {
  (void)in_sizes; (void)n_in; (void)out_size; (void)ws_size;
  const float*         x     = (const float*)d_in[0];
  const unsigned char* mask  = (const unsigned char*)d_in[1];  // bool8
  const float*         W_qkv = (const float*)d_in[2];
  const float*         b_qkv = (const float*)d_in[3];
  const float*         W_out = (const float*)d_in[4];
  const float*         b_out = (const float*)d_in[5];
  float* out = (float*)d_out;

  // Workspace carve (bf16 = unsigned short), ~92 MB total
  unsigned short* xb     = (unsigned short*)d_ws;                 // [8192,1024]
  unsigned short* wqkv_t = xb     + (size_t)8192 * 1024;          // [3072,1024]
  unsigned short* wout_t = wqkv_t + (size_t)3072 * 1024;          // [1024,1024]
  unsigned short* Qw     = wout_t + (size_t)1024 * 1024;          // [64,2048,64]
  unsigned short* Kw     = Qw     + (size_t)64 * 2048 * 64;
  unsigned short* Vtw    = Kw     + (size_t)64 * 2048 * 64;       // [64,64,2048]
  unsigned short* Ybuf   = Vtw    + (size_t)64 * 2048 * 64;       // [8192,1024]

  cvt_f32_bf16_kernel<<<4096, 256, 0, stream>>>(x, xb, 8192 * 1024 / 8);
  cvt_transpose_kernel<<<dim3(96, 32), 256, 0, stream>>>(W_qkv, wqkv_t, 1024, 3072);
  cvt_transpose_kernel<<<dim3(32, 32), 256, 0, stream>>>(W_out, wout_t, 1024, 1024);
  gemm_qkv_kernel<<<dim3(24, 64), 256, 0, stream>>>(xb, wqkv_t, b_qkv, Qw, Kw, Vtw);
  attn_kernel<<<dim3(16, 64), 256, 0, stream>>>(Qw, Kw, Vtw, mask, Ybuf);
  gemm_out_kernel<<<dim3(8, 64), 256, 0, stream>>>(Ybuf, wout_t, b_out, out);
}
